// TrajVAE_18691697672724
// MI455X (gfx1250) — compile-verified, matrix operand order checked by reading, not run on hardware
//
#include <hip/hip_runtime.h>
#include <hip/hip_bf16.h>

typedef __attribute__((ext_vector_type(16))) _Float16 v16h;
typedef __attribute__((ext_vector_type(8)))  _Float16 v8h;
typedef __attribute__((ext_vector_type(8)))  float    v8f;

#define SLEN 30
#define TSTEP 0.03f

// Output segment offsets (floats), tuple order:
// recons_traj, expert_traj, mu, log_var, out_label, traj_label
static constexpr long RECON_OFF  = 0;
static constexpr long EXPERT_OFF = 7864320L;    // + B*S*4
static constexpr long MU_OFF     = 15728640L;   // + B*S*4
static constexpr long LV_OFF     = 22282240L;   // + B*100
static constexpr long LAB_OFF    = 28835840L;   // + B*100
static constexpr long TLAB_OFF   = 28966912L;   // + B*2

struct Params {
  const float* expert; const float* init_state; const int* label; const float* eps;
  const float* emb_W; const float* emb_b;
  const float* enc_Wih; const float* enc_bih; const float* enc_Whh; const float* enc_bhh;
  const float* meanW[4]; const float* meanB[4];
  const float* lvW[4];   const float* lvB[4];
  const float* dec_emb_W; const float* dec_emb_b;
  const float* dec_Wih; const float* dec_bih; const float* dec_Whh; const float* dec_bhh;
  const float* h2c_W; const float* h2c_b;
  const float* initH_W; const float* initH_b;
  const float* labW[4]; const float* labB[4];
  float* out;
};

__device__ __forceinline__ float sigm(float x)  { return 1.0f / (1.0f + expf(-x)); }
__device__ __forceinline__ float lrelu(float x) { return x > 0.0f ? x : 0.01f * x; }

__device__ __forceinline__ v8f wmma16(v16h a, v16h b, v8f c) {
  return __builtin_amdgcn_wmma_f32_16x16x32_f16(false, a, false, b, (short)0, c,
                                                false, false);
}

// A fragment: 16x32 f16 tile from LDS, row-major, documented gfx1250 layout.
// lane<16: M=lane, K = kb..kb+7 and kb+16..kb+23 (kb=koff); lane>=16: kb=koff+8.
__device__ __forceinline__ v16h load_a_frag(const _Float16* A, int lda, int koff, int lane) {
  int m  = lane & 15;
  int kb = koff + ((lane & 16) ? 8 : 0);
  const _Float16* p = A + m * lda + kb;
  v8h lo = *(const v8h*)(p);        // ds_load_b128
  v8h hi = *(const v8h*)(p + 16);   // ds_load_b128
  v16h r;
#pragma unroll
  for (int i = 0; i < 8; ++i) { r[i] = lo[i]; r[i + 8] = hi[i]; }
  return r;
}

// B fragment: B[k][n] = W[n][k], W row-major [Nact][ldw] f32 in global, guarded.
__device__ __forceinline__ v16h load_b_frag_w(const float* W, int ldw, int Nact, int Kact,
                                              int nbase, int koff, int lane) {
  int n  = nbase + (lane & 15);
  int kb = koff + ((lane & 16) ? 8 : 0);
  v16h r;
#pragma unroll
  for (int i = 0; i < 16; ++i) {
    int k = kb + (i < 8 ? i : 8 + i);
    float w = (n < Nact && k < Kact) ? W[n * ldw + k] : 0.0f;
    r[i] = (_Float16)w;
  }
  return r;
}

// Encoder weights: K 0..65 -> Wih[n][k], 66..95 -> 0 pad, 96..159 -> Whh[n][k-96]
__device__ __forceinline__ v16h load_b_frag_enc(const float* Wih, const float* Whh,
                                                int nbase, int koff, int lane) {
  int n  = nbase + (lane & 15);
  int kb = koff + ((lane & 16) ? 8 : 0);
  v16h r;
#pragma unroll
  for (int i = 0; i < 16; ++i) {
    int k = kb + (i < 8 ? i : 8 + i);
    float w;
    if (k < 66)      w = Wih[n * 66 + k];
    else if (k < 96) w = 0.0f;
    else             w = Whh[n * 64 + (k - 96)];
    r[i] = (_Float16)w;
  }
  return r;
}

// Decoder weights: K 0..63 -> Wih, 64..127 -> Whh
__device__ __forceinline__ v16h load_b_frag_dec(const float* Wih, const float* Whh,
                                                int nbase, int koff, int lane) {
  int n  = nbase + (lane & 15);
  int kb = koff + ((lane & 16) ? 8 : 0);
  v16h r;
#pragma unroll
  for (int i = 0; i < 16; ++i) {
    int k = kb + (i < 8 ? i : 8 + i);
    float w = (k < 64) ? Wih[n * 64 + k] : Whh[n * 64 + (k - 64)];
    r[i] = (_Float16)w;
  }
  return r;
}

// One WMMA MLP layer: out[16xN] (f32, LDS stride 256) = lrelu later; bias fused in acc.
__device__ __forceinline__ void mlp_layer_wmma(const float* W, const float* bias,
                                               int Nact, int Kact, int kchunks,
                                               const _Float16* Ain, int a_off,
                                               float* Gout, int tile0, int tstep,
                                               int ntiles, int lane) {
  for (int nt = tile0; nt < ntiles; nt += tstep) {
    int ng = nt * 16;
    int n  = ng + (lane & 15);
    float bv = (n < Nact) ? bias[n] : 0.0f;
    v8f acc;
#pragma unroll
    for (int i = 0; i < 8; ++i) acc[i] = bv;
    for (int kc = 0; kc < kchunks; ++kc) {
      v16h a = load_a_frag(Ain, 192, a_off + kc * 32, lane);
      v16h b = load_b_frag_w(W, Kact, Nact, Kact, ng, kc * 32, lane);
      acc = wmma16(a, b, acc);
    }
    int mb = (lane & 16) ? 8 : 0;
#pragma unroll
    for (int r = 0; r < 8; ++r) Gout[(mb + r) * 256 + ng + (lane & 15)] = acc[r];
  }
}

__global__ __launch_bounds__(256) void trajvae_fused(Params p) {
  // LDS: activations (f16), gates (f32), cell state, bicycle state, controls
  __shared__ __attribute__((aligned(16))) _Float16 sA[16 * 192];  // 6 KB
  __shared__ float sG[16 * 256];                                  // 16 KB
  __shared__ float sC[16 * 64];                                   // 4 KB
  __shared__ float sSt[64];
  __shared__ float sCtrl[32];

  const int tid  = threadIdx.x;
  const int lane = tid & 31;
  const int wave = tid >> 5;
  const int b0   = blockIdx.x * 16;

  // ---- passthrough outputs: expert_traj copy + traj_label ----
  for (int i = tid; i < 16 * 120; i += 256)
    p.out[EXPERT_OFF + (long)b0 * 120 + i] = p.expert[(long)b0 * 120 + i];
  if (tid < 16)
    ((int*)(p.out + TLAB_OFF))[b0 + tid] = p.label[b0 + tid];

  // ---- init h0=c0=0 ----
  for (int i = tid; i < 16 * 64; i += 256) {
    int m = i >> 6, j = i & 63;
    sC[i] = 0.0f;
    sA[m * 192 + 96 + j] = (_Float16)0.0f;
  }
  __syncthreads();

  // ================= ENCODER LSTM (30 steps) =================
  {
    v16h Bf[2][5];           // per-wave persistent weight fragments (2 N-tiles x 5 K-chunks)
    float bias2[2];
#pragma unroll
    for (int j = 0; j < 2; ++j) {
      int ng = (wave * 2 + j) * 16;
      int n  = ng + (lane & 15);
      bias2[j] = p.enc_bih[n] + p.enc_bhh[n];
#pragma unroll
      for (int kc = 0; kc < 5; ++kc)
        Bf[j][kc] = load_b_frag_enc(p.enc_Wih, p.enc_Whh, ng, kc * 32, lane);
    }

    for (int t = 0; t < SLEN; ++t) {
      // build x_t = [onehot(label), rel @ embW^T + embB, 0-pad] into sA cols 0..95
      {
        int m = tid >> 4;
        int b = b0 + m;
        const float* tp = p.expert + (long)b * 120 + t * 4;
        float tx = tp[0], ty = tp[1];
        float r0, r1;
        if (t == 0) { r0 = tx; r1 = ty; }
        else        { r0 = tx - tp[-4]; r1 = ty - tp[-3]; }
        if (t + 1 < SLEN) __builtin_prefetch(tp + 4);   // global_prefetch_b8
        int lab = p.label[b];
        int c0  = (tid & 15) * 6;
#pragma unroll
        for (int i = 0; i < 6; ++i) {
          int col = c0 + i;
          float v;
          if (col < 2)       v = (lab == col) ? 1.0f : 0.0f;
          else if (col < 66) {
            int e = col - 2;
            v = r0 * p.emb_W[e * 2] + r1 * p.emb_W[e * 2 + 1] + p.emb_b[e];
          } else v = 0.0f;
          sA[m * 192 + col] = (_Float16)v;
        }
      }
      __syncthreads();

      // gates = [x|h] @ [Wih|Whh]^T + bias  (K=160, N=256; 10 WMMA per wave)
#pragma unroll
      for (int j = 0; j < 2; ++j) {
        v8f acc;
#pragma unroll
        for (int i = 0; i < 8; ++i) acc[i] = bias2[j];
#pragma unroll
        for (int kc = 0; kc < 5; ++kc) {
          v16h a = load_a_frag(sA, 192, kc * 32, lane);
          acc = wmma16(a, Bf[j][kc], acc);
        }
        int ng = (wave * 2 + j) * 16;
        int mb = (lane & 16) ? 8 : 0;
#pragma unroll
        for (int r = 0; r < 8; ++r) sG[(mb + r) * 256 + ng + (lane & 15)] = acc[r];
      }
      __syncthreads();

      // LSTM state update
#pragma unroll
      for (int q = 0; q < 4; ++q) {
        int e = tid + q * 256;
        int m = e >> 6, j = e & 63;
        float gi = sG[m * 256 + j],        gf = sG[m * 256 + 64 + j];
        float gg = sG[m * 256 + 128 + j],  go = sG[m * 256 + 192 + j];
        float c  = sigm(gf) * sC[m * 64 + j] + sigm(gi) * tanhf(gg);
        sC[m * 64 + j] = c;
        sA[m * 192 + 96 + j] = (_Float16)(sigm(go) * tanhf(c));
      }
      __syncthreads();
    }
  }

  // ================= LATENT: mu / log_var MLPs in parallel =================
  {
    const int half = wave >> 2;   // 0 = mean path, 1 = logvar path
    const int wsub = wave & 3;
    const float* W4[4]; const float* B4[4];
#pragma unroll
    for (int i = 0; i < 4; ++i) {
      W4[i] = half ? p.lvW[i] : p.meanW[i];
      B4[i] = half ? p.lvB[i] : p.meanB[i];
    }
    const int goff = half * 128;        // mu -> sG cols 0.., lv -> cols 128..
    const int aoff = half * 64;         // mu acts -> sA 0..63, lv acts -> 64..127

    // L1 reads hT (sA cols 96..159)
    mlp_layer_wmma(W4[0], B4[0], 64, 64, 2, sA, 96, sG + goff, wsub, 4, 4, lane);
    __syncthreads();
#pragma unroll
    for (int q = 0; q < 8; ++q) {       // lrelu -> f16 activations
      int e = tid + q * 256;  int m = e >> 7, col = e & 127;
      float v = (col < 64) ? sG[m * 256 + col] : sG[m * 256 + 128 + (col - 64)];
      sA[m * 192 + col] = (_Float16)lrelu(v);
    }
    __syncthreads();
    mlp_layer_wmma(W4[1], B4[1], 64, 64, 2, sA, aoff, sG + goff, wsub, 4, 4, lane);
    __syncthreads();
#pragma unroll
    for (int q = 0; q < 8; ++q) {
      int e = tid + q * 256;  int m = e >> 7, col = e & 127;
      float v = (col < 64) ? sG[m * 256 + col] : sG[m * 256 + 128 + (col - 64)];
      sA[m * 192 + col] = (_Float16)lrelu(v);
    }
    __syncthreads();
    mlp_layer_wmma(W4[2], B4[2], 64, 64, 2, sA, aoff, sG + goff, wsub, 4, 4, lane);
    __syncthreads();
#pragma unroll
    for (int q = 0; q < 8; ++q) {
      int e = tid + q * 256;  int m = e >> 7, col = e & 127;
      float v = (col < 64) ? sG[m * 256 + col] : sG[m * 256 + 128 + (col - 64)];
      sA[m * 192 + col] = (_Float16)lrelu(v);
    }
    __syncthreads();
    // L4: N=100 (7 tiles)
    mlp_layer_wmma(W4[3], B4[3], 100, 64, 2, sA, aoff, sG + goff, wsub, 4, 7, lane);
    __syncthreads();

    // z = eps * exp(0.5*logvar) + mu ; store mu/logvar; z (f16) -> sA cols 0..127
#pragma unroll
    for (int q = 0; q < 8; ++q) {
      int e = tid + q * 256;  int m = e >> 7, col = e & 127;
      long b = b0 + m;
      float zv = 0.0f;
      if (col < 100) {
        float mu = lrelu(sG[m * 256 + col]);
        float lv = lrelu(sG[m * 256 + 128 + col]);
        p.out[MU_OFF + b * 100 + col] = mu;
        p.out[LV_OFF + b * 100 + col] = lv;
        zv = p.eps[b * 100 + col] * expf(0.5f * lv) + mu;
      }
      sA[m * 192 + col] = (_Float16)zv;
    }
    __syncthreads();

    // label-MLP L1 (waves 0-3) || hd = z @ initH^T + b (waves 4-7), both K=100
    if (half == 0)
      mlp_layer_wmma(p.labW[0], p.labB[0], 64, 100, 4, sA, 0, sG,      wsub, 4, 4, lane);
    else
      mlp_layer_wmma(p.initH_W, p.initH_b, 64, 100, 4, sA, 0, sG + 64, wsub, 4, 4, lane);
    __syncthreads();
#pragma unroll
    for (int q = 0; q < 8; ++q) {
      int e = tid + q * 256;  int m = e >> 7, col = e & 127;
      if (col < 64) {
        sA[m * 192 + 128 + col] = (_Float16)lrelu(sG[m * 256 + col]);   // label act
      } else {
        int j = col - 64;
        float hd = sG[m * 256 + 64 + j];                                // h0 = c0 = hd
        sC[m * 64 + j] = hd;
        sA[m * 192 + 64 + j] = (_Float16)hd;
      }
    }
    if (tid < 64) sSt[tid] = p.init_state[(long)(b0 + (tid >> 2)) * 4 + (tid & 3)];
    __syncthreads();

    mlp_layer_wmma(p.labW[1], p.labB[1], 64, 64, 2, sA, 128, sG, wave, 8, 4, lane);
    __syncthreads();
#pragma unroll
    for (int q = 0; q < 4; ++q) {
      int e = tid + q * 256;  int m = e >> 6, j = e & 63;
      sA[m * 192 + 128 + j] = (_Float16)lrelu(sG[m * 256 + j]);
    }
    __syncthreads();
    mlp_layer_wmma(p.labW[2], p.labB[2], 64, 64, 2, sA, 128, sG, wave, 8, 4, lane);
    __syncthreads();
#pragma unroll
    for (int q = 0; q < 4; ++q) {
      int e = tid + q * 256;  int m = e >> 6, j = e & 63;
      sA[m * 192 + 128 + j] = (_Float16)lrelu(sG[m * 256 + j]);
    }
    __syncthreads();
    mlp_layer_wmma(p.labW[3], p.labB[3], 2, 64, 2, sA, 128, sG, wave, 8, 1, lane);
    __syncthreads();
    if (tid < 32) {
      int m = tid >> 1, j = tid & 1;
      p.out[LAB_OFF + (long)(b0 + m) * 2 + j] = lrelu(sG[m * 256 + j]);
    }
    __syncthreads();
  }

  // ================= DECODER LSTM + bicycle model (30 steps) =================
  {
    v16h Bf[2][4];
    float bias2[2];
#pragma unroll
    for (int j = 0; j < 2; ++j) {
      int ng = (wave * 2 + j) * 16;
      int n  = ng + (lane & 15);
      bias2[j] = p.dec_bih[n] + p.dec_bhh[n];
#pragma unroll
      for (int kc = 0; kc < 4; ++kc)
        Bf[j][kc] = load_b_frag_dec(p.dec_Wih, p.dec_Whh, ng, kc * 32, lane);
    }

    for (int t = 0; t < SLEN; ++t) {
      // x = st @ dec_emb_W^T + b  (4 -> 64, VALU) into sA cols 0..63
      {
        int m = tid >> 4;
        float s0 = sSt[m * 4], s1 = sSt[m * 4 + 1], s2 = sSt[m * 4 + 2], s3 = sSt[m * 4 + 3];
        int e0 = (tid & 15) * 4;
#pragma unroll
        for (int i = 0; i < 4; ++i) {
          int e = e0 + i;
          const float* w = p.dec_emb_W + e * 4;
          float v = p.dec_emb_b[e] + s0 * w[0] + s1 * w[1] + s2 * w[2] + s3 * w[3];
          sA[m * 192 + e] = (_Float16)v;
        }
      }
      __syncthreads();

      // gates (K=128, N=256; 8 WMMA per wave)
#pragma unroll
      for (int j = 0; j < 2; ++j) {
        v8f acc;
#pragma unroll
        for (int i = 0; i < 8; ++i) acc[i] = bias2[j];
#pragma unroll
        for (int kc = 0; kc < 4; ++kc) {
          v16h a = load_a_frag(sA, 192, kc * 32, lane);
          acc = wmma16(a, Bf[j][kc], acc);
        }
        int ng = (wave * 2 + j) * 16;
        int mb = (lane & 16) ? 8 : 0;
#pragma unroll
        for (int r = 0; r < 8; ++r) sG[(mb + r) * 256 + ng + (lane & 15)] = acc[r];
      }
      __syncthreads();

#pragma unroll
      for (int q = 0; q < 4; ++q) {
        int e = tid + q * 256;
        int m = e >> 6, j = e & 63;
        float gi = sG[m * 256 + j],       gf = sG[m * 256 + 64 + j];
        float gg = sG[m * 256 + 128 + j], go = sG[m * 256 + 192 + j];
        float c  = sigm(gf) * sC[m * 64 + j] + sigm(gi) * tanhf(gg);
        sC[m * 64 + j] = c;
        sA[m * 192 + 64 + j] = (_Float16)(sigm(go) * tanhf(c));
      }
      __syncthreads();

      // ctrl = h @ h2c^T + b (64 -> 2)
      if (tid < 32) {
        int m = tid & 15, j = tid >> 4;
        const float* w = p.h2c_W + j * 64;
        float s = p.h2c_b[j];
#pragma unroll 8
        for (int k = 0; k < 64; ++k) s += (float)sA[m * 192 + 64 + k] * w[k];
        sCtrl[j * 16 + m] = s;
      }
      __syncthreads();

      // bicycle dynamics + output
      if (tid < 16) {
        int m = tid; long b = b0 + m;
        float pedal = sCtrl[m];
        float steer = fminf(fmaxf(sCtrl[16 + m], -0.5f), 0.5f);
        float xt = sSt[m * 4], yt = sSt[m * 4 + 1], psi = sSt[m * 4 + 2], v = sSt[m * 4 + 3];
        float v1   = fminf(fmaxf(v + pedal * TSTEP, 0.0f), 10.0f);
        float pd   = fminf(fmaxf(v * tanf(steer) * (1.0f / 2.5f), -1.57f), 1.57f);
        float psi1 = pd * TSTEP + psi;
        float nx = xt + v1 * cosf(psi1) * TSTEP;
        float ny = yt + v1 * sinf(psi1) * TSTEP;
        sSt[m * 4] = nx; sSt[m * 4 + 1] = ny; sSt[m * 4 + 2] = psi1; sSt[m * 4 + 3] = v1;
        float* o = p.out + RECON_OFF + b * 120 + t * 4;
        o[0] = nx; o[1] = ny; o[2] = psi1; o[3] = v1;
      }
      __syncthreads();
    }
  }
}

extern "C" void kernel_launch(void* const* d_in, const int* in_sizes, int n_in,
                              void* d_out, int out_size, void* d_ws, size_t ws_size,
                              hipStream_t stream) {
  (void)in_sizes; (void)n_in; (void)d_ws; (void)ws_size; (void)out_size;
  Params p;
  p.expert     = (const float*)d_in[0];
  p.init_state = (const float*)d_in[1];
  p.label      = (const int*)  d_in[2];
  p.eps        = (const float*)d_in[3];
  p.emb_W      = (const float*)d_in[4];
  p.emb_b      = (const float*)d_in[5];
  p.enc_Wih    = (const float*)d_in[6];
  p.enc_bih    = (const float*)d_in[7];
  p.enc_Whh    = (const float*)d_in[8];
  p.enc_bhh    = (const float*)d_in[9];
  for (int k = 0; k < 4; ++k) p.meanW[k] = (const float*)d_in[10 + k];
  for (int k = 0; k < 4; ++k) p.meanB[k] = (const float*)d_in[14 + k];
  for (int k = 0; k < 4; ++k) p.lvW[k]   = (const float*)d_in[18 + k];
  for (int k = 0; k < 4; ++k) p.lvB[k]   = (const float*)d_in[22 + k];
  p.dec_emb_W = (const float*)d_in[26];
  p.dec_emb_b = (const float*)d_in[27];
  p.dec_Wih   = (const float*)d_in[28];
  p.dec_bih   = (const float*)d_in[29];
  p.dec_Whh   = (const float*)d_in[30];
  p.dec_bhh   = (const float*)d_in[31];
  p.h2c_W     = (const float*)d_in[32];
  p.h2c_b     = (const float*)d_in[33];
  p.initH_W   = (const float*)d_in[34];
  p.initH_b   = (const float*)d_in[35];
  for (int k = 0; k < 4; ++k) p.labW[k] = (const float*)d_in[36 + k];
  for (int k = 0; k < 4; ++k) p.labB[k] = (const float*)d_in[40 + k];
  p.out = (float*)d_out;

  trajvae_fused<<<dim3(65536 / 16), dim3(256), 0, stream>>>(p);
}
